// CausalSelfAttention_1967095021645
// MI455X (gfx1250) — compile-verified, matrix operand order checked by reading, not run on hardware
//
#include <hip/hip_runtime.h>
#include <hip/hip_bf16.h>
#include <cmath>

typedef _Float16 f16;
typedef __attribute__((ext_vector_type(16))) _Float16 v16h;
typedef __attribute__((ext_vector_type(8)))  _Float16 v8h;
typedef __attribute__((ext_vector_type(8)))  float    v8f;

#define DIMX  2048
#define NH    16
#define NKV   4
#define HDIM  128
#define RDIM  64
#define SEQ   2048
#define BATCH 2
#define MROWS (BATCH*SEQ)   // 4096
#define KVD   (NKV*HDIM)    // 512

// Async copy 16 bytes global -> LDS (CDNA5 GLOBAL_LOAD_ASYNC_TO_LDS_B128, ASYNCcnt).
// LDS generic pointers carry the LDS byte offset in addr[31:0] (aperture rule).
__device__ __forceinline__ void async_g2l_b128(void* lds_ptr, const void* gptr) {
  unsigned lds_off = (unsigned)(size_t)lds_ptr;
  unsigned long long ga = (unsigned long long)(size_t)gptr;
  asm volatile("global_load_async_to_lds_b128 %0, %1, off"
               :: "v"(lds_off), "v"(ga) : "memory");
}
__device__ __forceinline__ void wait_async0() {
  asm volatile("s_wait_asynccnt 0" ::: "memory");
}

// Load a 16x32 f16 A/B fragment from LDS per the CDNA5 WMMA layout:
// lane<16 holds K {0..7,16..23}; lane>=16 holds K {8..15,24..31}.
__device__ __forceinline__ v16h load_frag_lds(const f16* base, int row, int strideHalves, int lane) {
  const f16* p = base + row * strideHalves + ((lane & 16) ? 8 : 0);
  v8h lo = *(const v8h*)(p);
  v8h hi = *(const v8h*)(p + 16);
  v16h r;
#pragma unroll
  for (int i = 0; i < 8; ++i) { r[i] = lo[i]; r[i + 8] = hi[i]; }
  return r;
}

__global__ void cvt_f32_f16(const float* __restrict__ src, f16* __restrict__ dst, int n) {
  for (int i = blockIdx.x * blockDim.x + threadIdx.x; i < n; i += gridDim.x * blockDim.x)
    dst[i] = (f16)src[i];
}

// C[M,N] = A[M,K] * B[N,K]^T.  Block tile 128x64, BK=32, 8 waves, wave tile 32x32.
template<bool OUT_HALF>
__global__ __launch_bounds__(256) void gemm_nt_wmma(const f16* __restrict__ A,
                                                    const f16* __restrict__ Bw,
                                                    void* __restrict__ Cout,
                                                    int Ndim, int Kdim) {
  __shared__ alignas(16) f16 sA[128][40];
  __shared__ alignas(16) f16 sB[64][40];
  const int tid  = threadIdx.x;
  const int lane = tid & 31;
  const int wid  = tid >> 5;
  const int wm   = wid >> 1;   // 0..3
  const int wn   = wid & 1;    // 0..1
  const int m0   = blockIdx.y * 128;
  const int n0   = blockIdx.x * 64;

  v8f acc[2][2] = {};

  for (int k0 = 0; k0 < Kdim; k0 += 32) {
#pragma unroll
    for (int i = 0; i < 2; ++i) {             // A tile: 128x32 halves, async to LDS
      int ch = tid + i * 256;
      int r = ch >> 2, c = (ch & 3) << 3;
      async_g2l_b128(&sA[r][c], A + (size_t)(m0 + r) * Kdim + k0 + c);
    }
    {                                          // B tile: 64x32 halves, async to LDS
      int r = tid >> 2, c = (tid & 3) << 3;
      async_g2l_b128(&sB[r][c], Bw + (size_t)(n0 + r) * Kdim + k0 + c);
    }
    wait_async0();
    __syncthreads();

    v16h a[2], b[2];
#pragma unroll
    for (int t = 0; t < 2; ++t)
      a[t] = load_frag_lds(&sA[0][0], wm * 32 + t * 16 + (lane & 15), 40, lane);
#pragma unroll
    for (int t = 0; t < 2; ++t)
      b[t] = load_frag_lds(&sB[0][0], wn * 32 + t * 16 + (lane & 15), 40, lane);

#pragma unroll
    for (int tm = 0; tm < 2; ++tm)
#pragma unroll
      for (int tn = 0; tn < 2; ++tn)
        acc[tm][tn] = __builtin_amdgcn_wmma_f32_16x16x32_f16(
            false, a[tm], false, b[tn], (short)0, acc[tm][tn], false, false);
    __syncthreads();
  }

  const int rofs = (lane & 16) ? 8 : 0;
#pragma unroll
  for (int tm = 0; tm < 2; ++tm)
#pragma unroll
    for (int tn = 0; tn < 2; ++tn)
#pragma unroll
      for (int r = 0; r < 8; ++r) {
        int gm = m0 + wm * 32 + tm * 16 + r + rofs;
        int gn = n0 + wn * 32 + tn * 16 + (lane & 15);
        float v = acc[tm][tn][r];
        if (OUT_HALF) ((f16*)Cout)[(size_t)gm * Ndim + gn] = (f16)v;
        else          ((float*)Cout)[(size_t)gm * Ndim + gn] = v;
      }
}

// One wave per (token row, head): RMS-norm over 128 dims, RoPE on first 64, optional per-head gain.
__global__ __launch_bounds__(256) void rmsnorm_rope(const float* __restrict__ src,
                                                    f16* __restrict__ dst,
                                                    const float* __restrict__ gain,
                                                    int nHeads, float rope_base) {
  const int gw   = blockIdx.x * 8 + (threadIdx.x >> 5);
  const int lane = threadIdx.x & 31;
  const int h = gw % nHeads;
  const int m = gw / nHeads;
  const int s = m & (SEQ - 1);
  const int cols = nHeads * HDIM;
  const float* row = src + (size_t)m * cols + h * HDIM;

  float v0 = row[lane], v1 = row[lane + 32], v2 = row[lane + 64], v3 = row[lane + 96];
  float ss = v0 * v0 + v1 * v1 + v2 * v2 + v3 * v3;
#pragma unroll
  for (int off = 16; off >= 1; off >>= 1) ss += __shfl_xor(ss, off, 32);
  float rn = rsqrtf(ss * (1.0f / 128.0f) + 1.1920929e-07f);
  v0 *= rn; v1 *= rn; v2 *= rn; v3 *= rn;

  float inv_freq = powf(rope_base, -(float)(2 * lane) * (1.0f / (float)RDIM));
  float fr = (float)s * inv_freq;
  float c, sn;
  sincosf(fr, &sn, &c);
  float g = gain ? gain[h] : 1.0f;

  f16* drow = dst + (size_t)m * cols + h * HDIM;
  drow[lane]      = (f16)((v0 * c + v1 * sn) * g);
  drow[lane + 32] = (f16)((-v0 * sn + v1 * c) * g);
  drow[lane + 64] = (f16)(v2 * g);
  drow[lane + 96] = (f16)(v3 * g);
}

// Flash attention: block = (b, h, 64-row Q tile), 8 waves, K/V tiles of 32, online softmax.
__global__ __launch_bounds__(256) void flash_attn_wmma(const f16* __restrict__ Qh,
                                                       const f16* __restrict__ Kh,
                                                       const f16* __restrict__ Vh,
                                                       f16* __restrict__ Y) {
  __shared__ alignas(16) f16   sQ[64][136];
  __shared__ alignas(16) f16   sK[32][136];
  __shared__ alignas(16) f16   sVt[128][40];
  __shared__ alignas(16) float sS[64][33];
  __shared__ alignas(16) f16   sP[64][40];
  __shared__ float sCorr[64];
  __shared__ float sLfin[64];

  const int tid = threadIdx.x, lane = tid & 31, wid = tid >> 5;
  const int q0  = blockIdx.x * 64;
  const int h   = blockIdx.y;
  const int b   = blockIdx.z;
  const int kvh = h >> 2;
  const float scale = 0.08838834764831845f;  // 1/sqrt(128)

#pragma unroll
  for (int i = 0; i < 4; ++i) {  // Q tile 64x128, async to LDS
    int ch = tid + i * 256;
    int r = ch >> 4, c = (ch & 15) << 3;
    async_g2l_b128(&sQ[r][c], Qh + (size_t)(b * SEQ + q0 + r) * DIMX + h * HDIM + c);
  }

  const int wm = wid >> 1;  // S-tile / O-tile m index 0..3
  const int wn = wid & 1;
  v8f o[4] = {};
  float m_i = -INFINITY, l_i = 0.0f;

  const int nkb = (q0 >> 5) + 2;
  for (int kb = 0; kb < nkb; ++kb) {
    const int kbase = kb * 32;
#pragma unroll
    for (int i = 0; i < 2; ++i) {  // K tile 32x128, async to LDS
      int ch = tid + i * 256;
      int r = ch >> 4, c = (ch & 15) << 3;
      async_g2l_b128(&sK[r][c], Kh + (size_t)(b * SEQ + kbase + r) * KVD + kvh * HDIM + c);
    }
#pragma unroll
    for (int i = 0; i < 2; ++i) {  // V tile 32x128 stored transposed (scalar scatter)
      int ch = tid + i * 256;
      int r = ch >> 4, c = (ch & 15) << 3;
      v8h v = *(const v8h*)(Vh + (size_t)(b * SEQ + kbase + r) * KVD + kvh * HDIM + c);
#pragma unroll
      for (int j = 0; j < 8; ++j) sVt[c + j][r] = v[j];
    }
    wait_async0();       // covers Q (first iter) + K tile
    __syncthreads();

    // S tile (wm, wn): 16x16, K = 128 -> 4 WMMA steps
    v8f s_acc = {};
#pragma unroll
    for (int ks = 0; ks < 4; ++ks) {
      v16h aq = load_frag_lds(&sQ[0][ks * 32], wm * 16 + (lane & 15), 136, lane);
      v16h bk = load_frag_lds(&sK[0][ks * 32], wn * 16 + (lane & 15), 136, lane);
      s_acc = __builtin_amdgcn_wmma_f32_16x16x32_f16(false, aq, false, bk, (short)0, s_acc, false, false);
    }
    {
      const int rofs = (lane & 16) ? 8 : 0;
#pragma unroll
      for (int r = 0; r < 8; ++r)
        sS[wm * 16 + r + rofs][wn * 16 + (lane & 15)] = s_acc[r];
    }
    __syncthreads();

    // per-row online softmax, threads 0..63 own one query row each
    if (tid < 64) {
      const int qg = q0 + tid;
      float mx = m_i;
      float sc[32];
#pragma unroll
      for (int c = 0; c < 32; ++c) {
        int kg = kbase + c;
        float v = (kg <= qg) ? sS[tid][c] * scale : -INFINITY;
        sc[c] = v;
        mx = fmaxf(mx, v);
      }
      float corr = expf(m_i - mx);
      float rs = 0.0f;
#pragma unroll
      for (int c = 0; c < 32; ++c) {
        float p = (sc[c] == -INFINITY) ? 0.0f : expf(sc[c] - mx);
        rs += p;
        sP[tid][c] = (f16)p;
      }
      l_i = l_i * corr + rs;
      m_i = mx;
      sCorr[tid] = corr;
    }
    __syncthreads();

    // rescale accumulators and accumulate P @ V (O tiles: m = wm, n = wn*4 + j)
    const int rofs = (lane & 16) ? 8 : 0;
    v16h ap = load_frag_lds(&sP[0][0], wm * 16 + (lane & 15), 40, lane);
#pragma unroll
    for (int j = 0; j < 4; ++j) {
#pragma unroll
      for (int r = 0; r < 8; ++r)
        o[j][r] *= sCorr[wm * 16 + r + rofs];
      v16h bv = load_frag_lds(&sVt[0][0], (wn * 4 + j) * 16 + (lane & 15), 40, lane);
      o[j] = __builtin_amdgcn_wmma_f32_16x16x32_f16(false, ap, false, bv, (short)0, o[j], false, false);
    }
    __syncthreads();
  }

  if (tid < 64) sLfin[tid] = l_i;
  __syncthreads();

  const int rofs = (lane & 16) ? 8 : 0;
#pragma unroll
  for (int j = 0; j < 4; ++j)
#pragma unroll
    for (int r = 0; r < 8; ++r) {
      int row = wm * 16 + r + rofs;
      int d = (wn * 4 + j) * 16 + (lane & 15);
      float v = o[j][r] / sLfin[row];
      Y[(size_t)(b * SEQ + q0 + row) * DIMX + h * HDIM + d] = (f16)v;
    }
}

extern "C" void kernel_launch(void* const* d_in, const int* in_sizes, int n_in,
                              void* d_out, int out_size, void* d_ws, size_t ws_size,
                              hipStream_t stream) {
  (void)in_sizes; (void)n_in; (void)out_size; (void)ws_size;
  const float* x  = (const float*)d_in[0];
  const float* Wq = (const float*)d_in[1];
  const float* Wk = (const float*)d_in[2];
  const float* Wv = (const float*)d_in[3];
  const float* Wp = (const float*)d_in[4];
  const float* qg = (const float*)d_in[5];

  char* ws = (char*)d_ws;
  size_t off = 0;
  auto alloc = [&](size_t bytes) { char* p = ws + off; off += (bytes + 255) & ~(size_t)255; return p; };
  f16*   x_h  = (f16*)  alloc((size_t)MROWS * DIMX * 2);
  f16*   Wq_h = (f16*)  alloc((size_t)DIMX * DIMX * 2);
  f16*   Wk_h = (f16*)  alloc((size_t)KVD * DIMX * 2);
  f16*   Wv_h = (f16*)  alloc((size_t)KVD * DIMX * 2);
  f16*   Wp_h = (f16*)  alloc((size_t)DIMX * DIMX * 2);
  float* q_f  = (float*)alloc((size_t)MROWS * DIMX * 4);
  float* k_f  = (float*)alloc((size_t)MROWS * KVD * 4);
  f16*   q_h  = (f16*)  alloc((size_t)MROWS * DIMX * 2);
  f16*   k_h  = (f16*)  alloc((size_t)MROWS * KVD * 2);
  f16*   v_h  = (f16*)  alloc((size_t)MROWS * KVD * 2);
  f16*   y_h  = (f16*)  alloc((size_t)MROWS * DIMX * 2);

  cvt_f32_f16<<<512, 256, 0, stream>>>(x,  x_h,  MROWS * DIMX);
  cvt_f32_f16<<<512, 256, 0, stream>>>(Wq, Wq_h, DIMX * DIMX);
  cvt_f32_f16<<<256, 256, 0, stream>>>(Wk, Wk_h, KVD * DIMX);
  cvt_f32_f16<<<256, 256, 0, stream>>>(Wv, Wv_h, KVD * DIMX);
  cvt_f32_f16<<<512, 256, 0, stream>>>(Wp, Wp_h, DIMX * DIMX);

  gemm_nt_wmma<false><<<dim3(DIMX / 64, MROWS / 128), 256, 0, stream>>>(x_h, Wq_h, q_f, DIMX, DIMX);
  gemm_nt_wmma<false><<<dim3(KVD  / 64, MROWS / 128), 256, 0, stream>>>(x_h, Wk_h, k_f, KVD,  DIMX);
  gemm_nt_wmma<true ><<<dim3(KVD  / 64, MROWS / 128), 256, 0, stream>>>(x_h, Wv_h, v_h, KVD,  DIMX);

  // NTK rope base: S=2048 > TRAIN_SEQ=1024 -> base = 10000 * (S/TRAIN)^(RD/(RD-2))
  float rope_base = 10000.0f * powf(2.0f, (float)RDIM / (float)(RDIM - 2));
  rmsnorm_rope<<<(MROWS * NH)  / 8, 256, 0, stream>>>(q_f, q_h, qg,      NH,  rope_base);
  rmsnorm_rope<<<(MROWS * NKV) / 8, 256, 0, stream>>>(k_f, k_h, nullptr, NKV, rope_base);

  flash_attn_wmma<<<dim3(SEQ / 64, NH, BATCH), 256, 0, stream>>>(q_h, k_h, v_h, y_h);

  gemm_nt_wmma<false><<<dim3(DIMX / 64, MROWS / 128), 256, 0, stream>>>(y_h, Wp_h, (float*)d_out, DIMX, DIMX);
}
